// GraphSAGELayer_9861244911814
// MI455X (gfx1250) — matrix operand order repaired and checked by reading, not compiled, not run on hardware
//
#include <hip/hip_runtime.h>
#include <hip/hip_bf16.h>

typedef _Float16 half_t;
typedef __attribute__((ext_vector_type(16))) _Float16 v16h;
typedef __attribute__((ext_vector_type(8)))  _Float16 v8h;
typedef __attribute__((ext_vector_type(8)))  float    v8f;

#define NNODES 16384
#define NF     256
#define NSAMP  10
#define MAXC   128

// ---------------------------------------------------------------------------
// f32 -> f16 grid-stride convert
// ---------------------------------------------------------------------------
__global__ void cvt_f32_f16_kernel(const float* __restrict__ src,
                                   half_t* __restrict__ dst, int n) {
  for (int i = blockIdx.x * blockDim.x + threadIdx.x; i < n;
       i += gridDim.x * blockDim.x) {
    dst[i] = (half_t)src[i];
  }
}

// bias = b_self + b_neigh
__global__ void bias_sum_kernel(const float* __restrict__ b0,
                                const float* __restrict__ b1,
                                float* __restrict__ dst, int n) {
  int i = blockIdx.x * blockDim.x + threadIdx.x;
  if (i < n) dst[i] = b0[i] + b1[i];
}

// ---------------------------------------------------------------------------
// Neighbor sampling + mean aggregation. One 256-thread block per node.
// Coalesced float4 scan of the 64KB adjacency row; candidates pushed into an
// LDS list via ds atomics; wave-parallel top-10 select (shfl_xor argmax);
// 256-thread coalesced feature aggregate emitted as f16.
// ---------------------------------------------------------------------------
__device__ __forceinline__ float noise_hash(unsigned a, unsigned b) {
  unsigned h = a * 0x9E3779B1u ^ (b + 0x85EBCA6Bu);
  h ^= h >> 16; h *= 0x7FEB352Du;
  h ^= h >> 15; h *= 0x846CA68Bu;
  h ^= h >> 16;
  return (float)(h >> 8) * (1.0f / 16777216.0f);
}

__global__ void sample_agg_kernel(const float* __restrict__ x,
                                  const float* __restrict__ adj,
                                  half_t* __restrict__ aggh) {
  __shared__ int   s_count;
  __shared__ int   s_idx[MAXC];
  __shared__ float s_score[MAXC];
  __shared__ int   s_sel[NSAMP];
  __shared__ int   s_nsel;

  const int node = blockIdx.x;
  const int tid  = threadIdx.x;

  if (tid == 0) s_count = 0;
  __syncthreads();

  const float4* row = (const float4*)(adj + (size_t)node * NNODES);
  // 4096 float4 per row, 256 threads -> 16 iterations, fully coalesced
  for (int j4 = tid; j4 < NNODES / 4; j4 += 256) {
    __builtin_prefetch(row + j4 + 512, 0, 1);   // global_prefetch_b8
    float4 v = row[j4];
    float e[4] = {v.x, v.y, v.z, v.w};
#pragma unroll
    for (int k = 0; k < 4; ++k) {
      if (e[k] > 0.0f) {
        int col  = j4 * 4 + k;
        int slot = atomicAdd(&s_count, 1);      // ds_add_rtn_u32
        if (slot < MAXC) {
          s_idx[slot]   = col;
          s_score[slot] = noise_hash((unsigned)node, (unsigned)col);
        }
      }
    }
  }
  __syncthreads();

  // wave-parallel top-NSAMP selection on the first wave32
  if (tid < 32) {
    const int C = s_count < MAXC ? s_count : MAXC;
    int ns = 0;
#pragma unroll 1
    for (int s = 0; s < NSAMP; ++s) {
      float best = -1.0f;
      int   bidx = -1;
      for (int c = tid; c < C; c += 32) {
        float sc = s_score[c];
        if (sc > best) { best = sc; bidx = c; }
      }
#pragma unroll
      for (int o = 16; o > 0; o >>= 1) {        // wave32 argmax reduce
        float ob = __shfl_xor(best, o, 32);
        int   oi = __shfl_xor(bidx, o, 32);
        if (ob > best) { best = ob; bidx = oi; }
      }
      if (best < 0.0f) break;                   // uniform after reduction
      if (tid == 0) s_sel[ns] = s_idx[bidx];
      s_score[bidx] = -1.0f;                    // all lanes, same addr/value
      ++ns;
    }
    if (tid == 0) s_nsel = ns;
  }
  __syncthreads();

  const int   ns  = s_nsel;
  const float inv = (ns > 0) ? (1.0f / (float)ns) : 0.0f;
  float acc = 0.0f;
  for (int t = 0; t < ns; ++t) {
    acc += x[(size_t)s_sel[t] * NF + tid];      // coalesced row gathers
  }
  aggh[(size_t)node * NF + tid] = (half_t)(acc * inv);
}

// ---------------------------------------------------------------------------
// WMMA GEMM: out = xh @ Wself^T + aggh @ Wneigh^T + bias
// Wave32; each wave owns a 16-row strip, keeps A fragments (x + agg, K=256)
// in registers. The 8 waves of a block consume identical W tiles per N-step,
// so (Wself,Wneigh) 16-row tiles are double-buffered in LDS and filled with
// global_load_async_to_lds_b128 (ASYNCcnt), overlapping the next tile's DMA
// with the current tile's 16 WMMAs. B fragments read from LDS (ds_load).
// ---------------------------------------------------------------------------
#define TILE_BYTES 8192   // 16 rows x 256 halfs x 2B (contiguous in row-major W)

__device__ __forceinline__ void async_fill_tile(unsigned ldsbase, int buf,
                                                const half_t* __restrict__ Wsh,
                                                const half_t* __restrict__ Wnh,
                                                int nt, int t) {
  const unsigned t16  = (unsigned)t * 16u;
  const unsigned bufo = ldsbase + (unsigned)buf * (2u * TILE_BYTES);
  const half_t* srcS = Wsh + (size_t)nt * (TILE_BYTES / 2);
  const half_t* srcN = Wnh + (size_t)nt * (TILE_BYTES / 2);
#pragma unroll
  for (int seg = 0; seg < 2; ++seg) {
    unsigned voff = t16 + (unsigned)seg * 4096u;
    unsigned ldsS = bufo + voff;
    unsigned ldsN = bufo + TILE_BYTES + voff;
    asm volatile("global_load_async_to_lds_b128 %0, %1, %2"
                 :: "v"(ldsS), "v"(voff), "s"(srcS) : "memory");
    asm volatile("global_load_async_to_lds_b128 %0, %1, %2"
                 :: "v"(ldsN), "v"(voff), "s"(srcN) : "memory");
  }
}

__global__ void __launch_bounds__(256)
gemm_wmma_kernel(const half_t* __restrict__ xh,
                 const half_t* __restrict__ aggh,
                 const half_t* __restrict__ Wsh,
                 const half_t* __restrict__ Wnh,
                 const float*  __restrict__ bias,
                 float* __restrict__ out) {
  // [buf 2][mat 2][row 16][col 256] f16 = 32 KB
  __shared__ half_t wlds[2 * 2 * 16 * 256];

  const int lane = threadIdx.x & 31;
  const int wave = threadIdx.x >> 5;
  const int m0   = blockIdx.x * 128 + wave * 16;
  const unsigned ldsbase = (unsigned)(uintptr_t)&wlds[0];

  // ---- A fragments (contiguous k = 16h + s per lane half), register resident ----
  const int mrow = m0 + (lane & 15);
  const int koff = (lane & 16) ? 16 : 0;
  const half_t* xrow = xh   + (size_t)mrow * NF + koff;
  const half_t* arow = aggh + (size_t)mrow * NF + koff;

  v16h ax[8];
  v16h ag[8];
#pragma unroll
  for (int kc = 0; kc < 8; ++kc) {
    v8h lo = *(const v8h*)(xrow + kc * 32);
    v8h hi = *(const v8h*)(xrow + kc * 32 + 8);
    ax[kc] = __builtin_shufflevector(lo, hi, 0,1,2,3,4,5,6,7,8,9,10,11,12,13,14,15);
    v8h lo2 = *(const v8h*)(arow + kc * 32);
    v8h hi2 = *(const v8h*)(arow + kc * 32 + 8);
    ag[kc] = __builtin_shufflevector(lo2, hi2, 0,1,2,3,4,5,6,7,8,9,10,11,12,13,14,15);
  }

  // ---- B addressing (contiguous k = 16h + s, same order as A) from LDS tiles ----
  const int nlane = lane & 15;
  const int kb    = (lane & 16) ? 16 : 0;
  const int mstore = m0 + ((lane & 16) ? 8 : 0);

  // prologue: fill buffer 0 with N-tile 0
  async_fill_tile(ldsbase, 0, Wsh, Wnh, 0, threadIdx.x);
  asm volatile("s_wait_asynccnt 0" ::: "memory");
  __syncthreads();

  for (int nt = 0; nt < 16; ++nt) {
    const int buf = nt & 1;
    if (nt < 15) {
      async_fill_tile(ldsbase, buf ^ 1, Wsh, Wnh, nt + 1, threadIdx.x);
    }

    const half_t* bS = &wlds[((buf * 2 + 0) * 16 + nlane) * 256 + kb];
    const half_t* bN = &wlds[((buf * 2 + 1) * 16 + nlane) * 256 + kb];

    v8f acc = {};
#pragma unroll
    for (int kc = 0; kc < 8; ++kc) {
      v16h b = *(const v16h*)(bS + kc * 32);           // ds_load_b128 x2
      acc = __builtin_amdgcn_wmma_f32_16x16x32_f16(
          false, ax[kc], false, b, (short)0, acc, false, false);
    }
#pragma unroll
    for (int kc = 0; kc < 8; ++kc) {
      v16h b = *(const v16h*)(bN + kc * 32);
      acc = __builtin_amdgcn_wmma_f32_16x16x32_f16(
          false, ag[kc], false, b, (short)0, acc, false, false);
    }

    const int   o0  = nt * 16;
    const int   col = o0 + nlane;
    const float bo  = bias[col];
#pragma unroll
    for (int r = 0; r < 8; ++r) {
      out[(size_t)(mstore + r) * NF + col] = acc[r] + bo;
    }

    asm volatile("s_wait_asynccnt 0" ::: "memory");
    __syncthreads();   // next buffer filled; all waves done reading current
  }
}

// ---------------------------------------------------------------------------
// Launch
// ---------------------------------------------------------------------------
extern "C" void kernel_launch(void* const* d_in, const int* in_sizes, int n_in,
                              void* d_out, int out_size, void* d_ws, size_t ws_size,
                              hipStream_t stream) {
  const float* x      = (const float*)d_in[0];
  const float* adj    = (const float*)d_in[1];
  const float* W_self = (const float*)d_in[2];
  const float* b_self = (const float*)d_in[3];
  const float* W_neigh= (const float*)d_in[4];
  const float* b_neigh= (const float*)d_in[5];
  float* out = (float*)d_out;

  char* ws = (char*)d_ws;
  half_t* xh   = (half_t*)(ws);                                    // 8 MiB
  half_t* aggh = (half_t*)(ws + (size_t)8 * 1024 * 1024);          // 8 MiB
  half_t* Wsh  = (half_t*)(ws + (size_t)16 * 1024 * 1024);         // 128 KiB
  half_t* Wnh  = (half_t*)(ws + (size_t)16 * 1024 * 1024 + 128 * 1024);
  float*  bias = (float*) (ws + (size_t)16 * 1024 * 1024 + 256 * 1024);

  // stage f16 operands
  cvt_f32_f16_kernel<<<4096, 256, 0, stream>>>(x, xh, NNODES * NF);
  cvt_f32_f16_kernel<<<256, 256, 0, stream>>>(W_self,  Wsh, NF * NF);
  cvt_f32_f16_kernel<<<256, 256, 0, stream>>>(W_neigh, Wnh, NF * NF);
  bias_sum_kernel<<<1, 256, 0, stream>>>(b_self, b_neigh, bias, NF);

  // sample + aggregate (the HBM-bound 1 GB adjacency scan)
  sample_agg_kernel<<<NNODES, 256, 0, stream>>>(x, adj, aggh);

  // fused dual-GEMM + bias via WMMA, async W-tile staging
  gemm_wmma_kernel<<<NNODES / 128, 256, 0, stream>>>(xh, aggh, Wsh, Wnh, bias, out);
}
